// SingleOrderGNN_54211077210418
// MI455X (gfx1250) — matrix operand order repaired and checked by reading, not compiled, not run on hardware
//
#include <hip/hip_runtime.h>

typedef __bf16 bf16_t;
typedef bf16_t v8bf  __attribute__((ext_vector_type(8)));
typedef bf16_t v16bf __attribute__((ext_vector_type(16)));
typedef float  v8f   __attribute__((ext_vector_type(8)));

#define HCH 128  // channels (D == H == 128)

__device__ __forceinline__ unsigned short f2bf(float f) {
    union { float f; unsigned u; } v; v.f = f;
    unsigned u = v.u;
    unsigned r = (u + 0x7FFFu + ((u >> 16) & 1u)) >> 16;   // round-to-nearest-even
    return (unsigned short)r;
}

__device__ __forceinline__ void atomic_add_f32(float* p, float v) {
    // relaxed agent-scope fadd, result unused -> global_atomic_add_f32 (no return)
    (void)__hip_atomic_fetch_add(p, v, __ATOMIC_RELAXED, __HIP_MEMORY_SCOPE_AGENT);
}

// ---------------- degree / normalization ----------------
__global__ __launch_bounds__(256) void k_deg_init(float* deg, int n) {
    int i = blockIdx.x * 256 + threadIdx.x;
    if (i < n) deg[i] = 1.0f;                               // self loop
}

__global__ __launch_bounds__(256) void k_deg_edges(const int* dst, float* deg, int e) {
    int i = blockIdx.x * 256 + threadIdx.x;
    if (i < e) atomic_add_f32(&deg[dst[i]], 1.0f);
}

__global__ __launch_bounds__(256) void k_dinv(float* deg, int n) {
    int i = blockIdx.x * 256 + threadIdx.x;
    if (i < n) {
        float d = deg[i];
        d = d > 1e-12f ? d : 1e-12f;
        deg[i] = rsqrtf(d);
    }
}

// ---------------- fp32 -> bf16 converts ----------------
// weights: 128x128, stored TRANSPOSED (col-major, [n][k]) so WMMA B fragments
// are contiguous per lane.
__global__ __launch_bounds__(256) void k_conv_w(const float* __restrict__ W,
                                                unsigned short* __restrict__ Wt) {
    int idx = blockIdx.x * 256 + threadIdx.x;      // 16384 elements
    int k = idx >> 7, n = idx & 127;
    Wt[n * HCH + k] = f2bf(W[k * HCH + n]);
}

// activations: N*128 elements, vectorized x4
__global__ __launch_bounds__(256) void k_conv_act(const float* __restrict__ src,
                                                  unsigned short* __restrict__ dst,
                                                  int n4) {
    int i = blockIdx.x * 256 + threadIdx.x;
    if (i < n4) {
        float4 v = ((const float4*)src)[i];
        ushort4 o;
        o.x = f2bf(v.x); o.y = f2bf(v.y); o.z = f2bf(v.z); o.w = f2bf(v.w);
        ((ushort4*)dst)[i] = o;
    }
}

// ---------------- bf16 WMMA GEMM with fused epilogue ----------------
// C[n x 128] = A[n x 128] * W[128 x 128] (+bias)
// Epilogue can emit any of:  Cf  = fp32 result ("t" for the gather)
//                            Cbf = bf16 result (next GEMM's A)
//                            hs  = result * dinv[row]^2 (self-loop init of accum)
__global__ __launch_bounds__(256) void k_gemm_bf16(const bf16_t* __restrict__ A,
                                                   const bf16_t* __restrict__ Wt,
                                                   const float* __restrict__ bias,
                                                   float* __restrict__ Cf,
                                                   unsigned short* __restrict__ Cbf,
                                                   float* __restrict__ hs,
                                                   const float* __restrict__ dinv,
                                                   int nrows, int add_bias) {
    __shared__ bf16_t ldsW[HCH * HCH];   // 32 KB of 320 KB WGP LDS
    {   // cooperative 128b stage of the transposed weights
        const uint4* g = (const uint4*)Wt;
        uint4* l = (uint4*)ldsW;
        #pragma unroll
        for (int i = threadIdx.x; i < (HCH * HCH * 2) / 16; i += 256) l[i] = g[i];
    }
    __syncthreads();

    const int wave = threadIdx.x >> 5;
    const int lane = threadIdx.x & 31;
    const int half = lane >> 4;
    const int col0 = lane & 15;
    const int row_base = blockIdx.x * 128 + wave * 16;

    // A-fragment source row for this lane (ISA: lane holds M = lane%16)
    int r = row_base + col0;
    int rc = r < nrows ? r : (nrows - 1);   // clamp: keep EXEC all-ones for WMMA

    v8f acc[8] = {};

    #pragma unroll
    for (int kc = 0; kc < 4; ++kc) {
        const int kb = kc * 32;
        // A 16x32 bf16 layout: lanes<16 -> K = kb+[0..7], kb+16+[0..7]
        //                      lanes>=16 -> K = kb+8+[0..7], kb+24+[0..7]
        const bf16_t* ap = A + (size_t)rc * HCH + kb + 8 * half;
        v8bf alo = *(const v8bf*)ap;
        v8bf ahi = *(const v8bf*)(ap + 16);
        v16bf av = __builtin_shufflevector(alo, ahi,
            0,1,2,3,4,5,6,7,8,9,10,11,12,13,14,15);
        #pragma unroll
        for (int ct = 0; ct < 8; ++ct) {
            // B 32x16 bf16 layout: lanes<16 -> K = kb+[0..7], kb+16+[0..7]; lanes>=16 -> K = kb+8+[0..7], kb+24+[0..7]
            const bf16_t* bp = &ldsW[(ct * 16 + col0) * HCH + kb + 8 * half];
            v8bf blo = *(const v8bf*)bp;
            v8bf bhi = *(const v8bf*)(bp + 16);
            v16bf bv = __builtin_shufflevector(blo, bhi,
                0,1,2,3,4,5,6,7,8,9,10,11,12,13,14,15);
            acc[ct] = __builtin_amdgcn_wmma_f32_16x16x32_bf16(
                false, av, false, bv, (short)0, acc[ct], false, false);
        }
    }

    // C layout: VGPR j -> row = row_base + 8*half + j, col = ct*16 + lane%16
    const int rbase = row_base + 8 * half;

    // per-j self-loop weight dinv[row]^2 (shared across all col-tiles)
    float dw[8];
    if (hs) {
        #pragma unroll
        for (int j = 0; j < 8; ++j) {
            int row = rbase + j;
            float dv = (row < nrows) ? dinv[row] : 0.0f;
            dw[j] = dv * dv;
        }
    }

    #pragma unroll
    for (int ct = 0; ct < 8; ++ct) {
        float bv = add_bias ? bias[ct * 16 + col0] : 0.0f;
        #pragma unroll
        for (int j = 0; j < 8; ++j) {
            int row = rbase + j;
            if (row < nrows) {
                size_t idx = (size_t)row * HCH + ct * 16 + col0;
                float val = acc[ct][j] + bv;
                if (Cf)  Cf[idx] = val;
                if (Cbf) Cbf[idx] = f2bf(val);
                if (hs)  hs[idx] = val * dw[j];
            }
        }
    }
}

// ---------------- edge scatter: wave per edge, lane-vectorized float4 ----------------
__global__ __launch_bounds__(256) void k_edge_scatter(const int* __restrict__ src,
                                                      const int* __restrict__ dst,
                                                      const float* __restrict__ dinv,
                                                      const float* __restrict__ t,
                                                      float* __restrict__ hs, int e) {
    int edge = blockIdx.x * 8 + (threadIdx.x >> 5);
    if (edge >= e) return;
    int lane = threadIdx.x & 31;
    int s = src[edge], d = dst[edge];
    float w = dinv[s] * dinv[d];
    int c = lane * 4;
    float4 v = *(const float4*)(t + (size_t)s * HCH + c);
    float* p = hs + (size_t)d * HCH + c;
    atomic_add_f32(p + 0, v.x * w);
    atomic_add_f32(p + 1, v.y * w);
    atomic_add_f32(p + 2, v.z * w);
    atomic_add_f32(p + 3, v.w * w);
}

// ---------------- bias + ReLU, nullable fp32 / bf16 outputs ----------------
__global__ __launch_bounds__(256) void k_finalize(const float* __restrict__ hsum,
                                                  const float* __restrict__ bias,
                                                  float* __restrict__ hf,
                                                  unsigned short* __restrict__ hb,
                                                  int n4) {
    int i = blockIdx.x * 256 + threadIdx.x;       // n*32 threads, float4 each
    if (i < n4) {
        int c = (i & 31) * 4;                     // channel within row
        float4 v = ((const float4*)hsum)[i];
        v.x = fmaxf(v.x + bias[c + 0], 0.0f);
        v.y = fmaxf(v.y + bias[c + 1], 0.0f);
        v.z = fmaxf(v.z + bias[c + 2], 0.0f);
        v.w = fmaxf(v.w + bias[c + 3], 0.0f);
        if (hf) ((float4*)hf)[i] = v;
        if (hb) {
            ushort4 o;
            o.x = f2bf(v.x); o.y = f2bf(v.y); o.z = f2bf(v.z); o.w = f2bf(v.w);
            ((ushort4*)hb)[i] = o;
        }
    }
}

// ---------------- prediction head: sigmoid(h . Wp + bp), wave per node ----------------
__global__ __launch_bounds__(256) void k_pred(const float* __restrict__ h,
                                              const float* __restrict__ Wp,
                                              const float* __restrict__ bp,
                                              float* __restrict__ out, int n) {
    int node = blockIdx.x * 8 + (threadIdx.x >> 5);
    if (node >= n) return;
    int lane = threadIdx.x & 31;
    float4 hv = *(const float4*)(h + (size_t)node * HCH + lane * 4);
    float4 wv = *(const float4*)(Wp + lane * 4);
    float s = hv.x * wv.x + hv.y * wv.y + hv.z * wv.z + hv.w * wv.w;
    #pragma unroll
    for (int off = 16; off >= 1; off >>= 1) s += __shfl_xor(s, off, 32);
    if (lane == 0) out[node] = 1.0f / (1.0f + __expf(-(s + bp[0])));
}

extern "C" void kernel_launch(void* const* d_in, const int* in_sizes, int n_in,
                              void* d_out, int out_size, void* d_ws, size_t ws_size,
                              hipStream_t stream) {
    const int N = in_sizes[0] / HCH;
    const int E = in_sizes[1] / 2;

    const float* x      = (const float*)d_in[0];
    const int*   ei     = (const int*)d_in[1];
    const float* W_init = (const float*)d_in[2];
    const float* b_init = (const float*)d_in[3];
    const float* W1     = (const float*)d_in[4];
    const float* b1     = (const float*)d_in[5];
    const float* W2     = (const float*)d_in[6];
    const float* b2     = (const float*)d_in[7];
    const float* Wp     = (const float*)d_in[8];
    const float* bp     = (const float*)d_in[9];
    const int* e_src = ei;
    const int* e_dst = ei + E;

    // workspace carve-up (256B aligned)
    char* ws = (char*)d_ws;
    size_t o = 0;
    auto al = [](size_t v) { return (v + 255) & ~(size_t)255; };
    float* dinv = (float*)(ws + o);                 o = al(o + (size_t)N * 4);
    float* B1   = (float*)(ws + o);                 o = al(o + (size_t)N * HCH * 4);  // t
    float* B2   = (float*)(ws + o);                 o = al(o + (size_t)N * HCH * 4);  // accum / h
    unsigned short* XB = (unsigned short*)(ws + o); o = al(o + (size_t)N * HCH * 2);  // bf16 act (in)
    unsigned short* HB = (unsigned short*)(ws + o); o = al(o + (size_t)N * HCH * 2);  // bf16 act (out)
    unsigned short* Wt0 = (unsigned short*)(ws + o); o = al(o + HCH * HCH * 2);
    unsigned short* Wt1 = (unsigned short*)(ws + o); o = al(o + HCH * HCH * 2);
    unsigned short* Wt2 = (unsigned short*)(ws + o); o = al(o + HCH * HCH * 2);

    const int gN    = (N + 255) / 256;
    const int gE    = (E + 255) / 256;
    const int gEl   = (N * 32 + 255) / 256;     // elementwise float4 passes
    const int gGemm = (N + 127) / 128;
    const int gWav  = (N + 7) / 8;              // wave-per-node
    const int gEdg  = (E + 7) / 8;              // wave-per-edge

    // normalization: deg = 1 + sum(dst==i); dinv = rsqrt(max(deg,1e-12))
    k_deg_init<<<gN, 256, 0, stream>>>(dinv, N);
    k_deg_edges<<<gE, 256, 0, stream>>>(e_dst, dinv, E);
    k_dinv<<<gN, 256, 0, stream>>>(dinv, N);

    // weights -> transposed bf16
    k_conv_w<<<64, 256, 0, stream>>>(W_init, Wt0);
    k_conv_w<<<64, 256, 0, stream>>>(W1, Wt1);
    k_conv_w<<<64, 256, 0, stream>>>(W2, Wt2);

    // h0 = x @ W_init + b_init   (consumed only as bf16 -> emit bf16 directly)
    k_conv_act<<<gEl, 256, 0, stream>>>(x, XB, N * 32);
    k_gemm_bf16<<<gGemm, 256, 0, stream>>>((const bf16_t*)XB, (const bf16_t*)Wt0,
                                           b_init, nullptr, HB, nullptr, nullptr, N, 1);

    // layer 1: t = h0 @ W1 ; fused self-loop init ; scatter ; relu(+b1) -> bf16 only
    k_gemm_bf16<<<gGemm, 256, 0, stream>>>((const bf16_t*)HB, (const bf16_t*)Wt1,
                                           nullptr, B1, nullptr, B2, dinv, N, 0);
    k_edge_scatter<<<gEdg, 256, 0, stream>>>(e_src, e_dst, dinv, B1, B2, E);
    k_finalize<<<gEl, 256, 0, stream>>>(B2, b1, nullptr, XB, N * 32);

    // layer 2: t = h1 @ W2 ; fused self-loop init ; scatter ; relu(+b2) -> fp32 only
    k_gemm_bf16<<<gGemm, 256, 0, stream>>>((const bf16_t*)XB, (const bf16_t*)Wt2,
                                           nullptr, B1, nullptr, B2, dinv, N, 0);
    k_edge_scatter<<<gEdg, 256, 0, stream>>>(e_src, e_dst, dinv, B1, B2, E);
    k_finalize<<<gEl, 256, 0, stream>>>(B2, b2, B1, nullptr, N * 32);

    // pred = sigmoid(h2 @ W_pred + b_pred)
    k_pred<<<gWav, 256, 0, stream>>>(B1, Wp, bp, (float*)d_out, N);
}
